// MPO_LSTMCell_all_62294205661397
// MI455X (gfx1250) — matrix operand order repaired and checked by reading, not compiled, not run on hardware
//
#include <hip/hip_runtime.h>
#include <hip/hip_bf16.h>
#include <math.h>

// MPO/TT LSTM cell for gfx1250 (MI455X, wave32, WMMA).
// Chain of 4 GEMMs per TT apply; k=2/1/0 run on v_wmma_f32_16x16x32_f16.

typedef _Float16 half_t;
typedef __attribute__((ext_vector_type(16))) _Float16 v16h;
typedef __attribute__((ext_vector_type(8)))  _Float16 v8h;
typedef __attribute__((ext_vector_type(4)))  _Float16 v4h;
typedef __attribute__((ext_vector_type(8)))  float    v8f;

#define DEV __device__ __forceinline__

// ------------------------------------------------------------------
// Core rearrangement: f32 TT cores -> f16, transposed to [n][k] with
// output index major in n so WMMA epilogue scatters are r-contiguous.
//   src flat (per reference _core_slices, kernel_w/u):
//     [0,4096)        G3[i3][r3*8+o3]
//     [4096,266240)   G2[(i2*64+r3)][(r2*8+o2)]
//     [266240,528384) G1[(i1*64+r2)][(r1*8+o1)]
//     [528384,544768) G0[(i0*64+r1)][o0]
//   dst flat (f16):
//     [0,4096)        G3R[o3][r3][i3]
//     [4096,266240)   G2T[(o2*64+r2)][(i2*64+r3)]
//     [266240,528384) G1T[(o1*64+r1)][(i1*64+r2)]
//     [528384,544768) G0T[o0][(i0*64+r1)]
// ------------------------------------------------------------------
__global__ void rearrange_cores(const float* __restrict__ src,
                                half_t* __restrict__ dst) {
    int idx = blockIdx.x * 256 + threadIdx.x;
    if (idx >= 544768) return;
    float v;
    if (idx < 4096) {
        int o3 = idx >> 9, rem = idx & 511;
        int r3 = rem >> 3, i3 = rem & 7;
        v = src[i3 * 512 + r3 * 8 + o3];
    } else if (idx < 266240) {
        int t = idx - 4096;
        int n = t >> 9, k = t & 511;
        int o2 = n >> 6, r2 = n & 63;
        v = src[4096 + k * 512 + r2 * 8 + o2];
    } else if (idx < 528384) {
        int t = idx - 266240;
        int n = t >> 9, k = t & 511;
        int o1 = n >> 6, r1 = n & 63;
        v = src[266240 + k * 512 + r1 * 8 + o1];
    } else {
        int t = idx - 528384;
        int o0 = t >> 9, k = t & 511;
        v = src[528384 + k * 32 + o0];
    }
    dst[idx] = (half_t)v;
}

// ------------------------------------------------------------------
// Stage k=3 ("expand"): T3[(o3,b,i0,i1)][(i2*64+r3)] =
//   sum_i3 x[b0+b][(i0*8+i1)*64 + i2*8 + i3] * G3R[o3][r3][i3]
// K=8 -> memory-bound VALU kernel; writes the exact row-major A layout
// the k=2 WMMA GEMM consumes (fully coalesced f16 stores).
// ------------------------------------------------------------------
__global__ void tt_expand(const float* __restrict__ X,
                          const half_t* __restrict__ G3R,
                          half_t* __restrict__ out, int b0, int lbc) {
    int idx = blockIdx.x * 256 + threadIdx.x;   // [0, 512*Bc*512)
    int col = idx & 511;                        // i2*64 + r3
    int row = idx >> 9;                         // (o3,b,i0,i1)
    int i1 = row & 7, i0 = (row >> 3) & 7;
    int b  = (row >> 6) & ((1 << lbc) - 1);
    int o3 = row >> (6 + lbc);
    int r3 = col & 63, i2 = col >> 6;
    const float*  xp = X + (size_t)(b0 + b) * 4096 + (i0 * 8 + i1) * 64 + i2 * 8;
    const half_t* gp = G3R + o3 * 512 + r3 * 8;
    float acc = 0.f;
#pragma unroll
    for (int i3 = 0; i3 < 8; ++i3) acc += xp[i3] * (float)gp[i3];
    out[idx] = (half_t)acc;
}

// ---- WMMA fragment gathers from padded LDS tiles (stride 40 halves) ----
DEV v16h load_fragA(const half_t* ls, int mbase, int lane) {
    // A 16x32 f16: lanes 0-15 hold M=lane, K {0..7,16..23}; lanes 16-31 M=lane-16, K {8..15,24..31}
    int mr = mbase + (lane & 15);
    int k0 = (lane >> 4) * 8;
    v8h lo = *(const v8h*)(ls + mr * 40 + k0);
    v8h hi = *(const v8h*)(ls + mr * 40 + k0 + 16);
    return __builtin_shufflevector(lo, hi, 0,1,2,3,4,5,6,7,8,9,10,11,12,13,14,15);
}
DEV v16h load_fragB(const half_t* ls, int nbase, int lane) {
    // B 32x16 f16: lanes 0-15 N=lane, K {0..7,16..23}; lanes 16-31 N=lane-16, K {8..15,24..31}
    int nr = nbase + (lane & 15);
    int k0 = (lane >> 4) * 8;
    v8h lo = *(const v8h*)(ls + nr * 40 + k0);
    v8h hi = *(const v8h*)(ls + nr * 40 + k0 + 16);
    return __builtin_shufflevector(lo, hi, 0,1,2,3,4,5,6,7,8,9,10,11,12,13,14,15);
}

// ------------------------------------------------------------------
// WMMA GEMM: C[M x N] = A[M x 512](f16,row-major) * Bt^T, where Bt is the
// pre-transposed core [N x 512] (f16, contiguous K per output column).
// STEP selects N and the permutation folded into the epilogue store:
//   STEP 2: N=512, C[(o3,b,i0,i1)][(o2,r2)] -> bufB[(o2,o3,b,i0)][(i1,r2)]
//   STEP 3: N=512, C[(o2,o3,b,i0)][(o1,r1)] -> bufA[(o1,o2,o3,b)][(i0,r1)]
//   STEP 4: N=32,  C[(o1,o2,o3,b)][o0]      -> zN row-major f32
// Block 128xBN, 8 waves, K=512 in 16 steps of 32, double-buffered LDS.
// ------------------------------------------------------------------
template <int STEP>
__global__ __launch_bounds__(256)
void tt_gemm(const half_t* __restrict__ A, const half_t* __restrict__ Bt,
             void* __restrict__ Cout, int lbc) {
    constexpr int BN  = (STEP == 4) ? 32 : 128;
    constexpr int WTM = (STEP == 4) ? 1 : 2;   // 16-row tiles per wave
    constexpr int WTN = (STEP == 4) ? 2 : 4;   // 16-col tiles per wave
    constexpr int LDA = 40;                    // padded LDS row stride (halves)

    __shared__ alignas(16) half_t lsA[2][128 * LDA];
    __shared__ alignas(16) half_t lsB[2][BN * LDA];

    const int tid  = threadIdx.x;
    const int wave = tid >> 5;
    const int lane = tid & 31;
    const int waveM = (STEP == 4) ? wave : (wave >> 1);
    const int waveN = (STEP == 4) ? 0    : (wave & 1);
    const int mb = blockIdx.x * 128;
    const int nb = blockIdx.y * BN;

    v8f acc[WTM][WTN];
#pragma unroll
    for (int i = 0; i < WTM; ++i)
#pragma unroll
        for (int j = 0; j < WTN; ++j) { v8f z = {}; acc[i][j] = z; }

    v8h ra[2]; v8h rb[2]; v4h rb4;

    auto gload = [&](int kt) {
#pragma unroll
        for (int p = 0; p < 2; ++p) {
            int c = tid + 256 * p, r = c >> 2, off = (c & 3) * 8;
            ra[p] = *(const v8h*)(A + (size_t)(mb + r) * 512 + kt * 32 + off);
        }
        if constexpr (BN == 128) {
#pragma unroll
            for (int p = 0; p < 2; ++p) {
                int c = tid + 256 * p, r = c >> 2, off = (c & 3) * 8;
                rb[p] = *(const v8h*)(Bt + (size_t)(nb + r) * 512 + kt * 32 + off);
            }
        } else {
            int r = tid >> 3, off = (tid & 7) * 4;
            rb4 = *(const v4h*)(Bt + (size_t)(nb + r) * 512 + kt * 32 + off);
        }
    };
    auto sstore = [&](int buf) {
#pragma unroll
        for (int p = 0; p < 2; ++p) {
            int c = tid + 256 * p, r = c >> 2, off = (c & 3) * 8;
            *(v8h*)&lsA[buf][r * LDA + off] = ra[p];
        }
        if constexpr (BN == 128) {
#pragma unroll
            for (int p = 0; p < 2; ++p) {
                int c = tid + 256 * p, r = c >> 2, off = (c & 3) * 8;
                *(v8h*)&lsB[buf][r * LDA + off] = rb[p];
            }
        } else {
            *(v4h*)&lsB[buf][(tid >> 3) * LDA + (tid & 7) * 4] = rb4;
        }
    };

    gload(0);
    sstore(0);
    __syncthreads();

    for (int kt = 0; kt < 16; ++kt) {
        const int cur = kt & 1;
        const bool more = (kt + 1) < 16;
        if (more) gload(kt + 1);

        v16h fa[WTM], fb[WTN];
#pragma unroll
        for (int i = 0; i < WTM; ++i)
            fa[i] = load_fragA(lsA[cur], waveM * (WTM * 16) + i * 16, lane);
#pragma unroll
        for (int j = 0; j < WTN; ++j)
            fb[j] = load_fragB(lsB[cur], waveN * (WTN * 16) + j * 16, lane);
#pragma unroll
        for (int i = 0; i < WTM; ++i)
#pragma unroll
            for (int j = 0; j < WTN; ++j)
                acc[i][j] = __builtin_amdgcn_wmma_f32_16x16x32_f16(
                    false, fa[i], false, fb[j], (short)0, acc[i][j], false, false);

        if (more) sstore(cur ^ 1);
        __syncthreads();
    }

    // Epilogue: permutation folded into the store address.
    const int Bm = (1 << lbc) - 1;
#pragma unroll
    for (int i = 0; i < WTM; ++i) {
#pragma unroll
        for (int j = 0; j < WTN; ++j) {
            int Rb = mb + waveM * (WTM * 16) + i * 16 + ((lane >> 4) << 3);
            int Cb = nb + waveN * (WTN * 16) + j * 16 + (lane & 15);
#pragma unroll
            for (int v = 0; v < 8; ++v) {
                int row = Rb + v;
                int col = Cb;
                float val = acc[i][j][v];
                if constexpr (STEP == 2) {
                    int i1 = row & 7, i0 = (row >> 3) & 7;
                    int b  = (row >> 6) & Bm, o3 = row >> (6 + lbc);
                    int r2 = col & 63, o2 = col >> 6;
                    size_t dst = (size_t)(((o2 * 8 + o3) << lbc) + b) * 4096
                               + i0 * 512 + i1 * 64 + r2;
                    ((half_t*)Cout)[dst] = (half_t)val;
                } else if constexpr (STEP == 3) {
                    int i0 = row & 7, b = (row >> 3) & Bm;
                    int o3 = (row >> (3 + lbc)) & 7, o2 = row >> (6 + lbc);
                    int r1 = col & 63, o1 = col >> 6;
                    size_t dst = (size_t)((((o1 * 8 + o2) * 8 + o3) << lbc) + b) * 512
                               + i0 * 64 + r1;
                    ((half_t*)Cout)[dst] = (half_t)val;
                } else {
                    ((float*)Cout)[(size_t)row * 32 + col] = val;
                }
            }
        }
    }
}

// ------------------------------------------------------------------
// Fused LSTM gates per batch chunk.
// zN layout: [(opart=o1*64+o2*8+o3)][b_local][o0], gate g of unit u sits at
//   ((u&511)*Bc + b)*32 + g*8 + (u>>9).
// ------------------------------------------------------------------
__global__ void lstm_gates(const float* __restrict__ zw, const float* __restrict__ zu,
                           const float* __restrict__ bias, const float* __restrict__ c_tm1,
                           float* __restrict__ h_out, float* __restrict__ c_out,
                           int b0, int lbc) {
    int idx = blockIdx.x * 256 + threadIdx.x;   // [0, Bc*4096)
    int u  = idx & 4095;
    int bl = idx >> 12;
    int bg = b0 + bl;
    int Bc = 1 << lbc;
    int base = ((u & 511) * Bc + bl) * 32 + (u >> 9);
    float z[4];
#pragma unroll
    for (int g = 0; g < 4; ++g) {
        int a = base + g * 8;
        z[g] = zw[a] + zu[a] + bias[g * 4096 + u];
    }
    auto hs = [](float x) { return fminf(fmaxf(0.2f * x + 0.5f, 0.f), 1.f); };
    float gi = hs(z[0]), gf = hs(z[1]), go = hs(z[3]);
    float cp = c_tm1[(size_t)bg * 4096 + u];
    float c  = gf * cp + gi * tanhf(z[2]);
    float h  = go * tanhf(c);
    h_out[(size_t)bg * 4096 + u] = h;
    c_out[(size_t)bg * 4096 + u] = c;
}

// ------------------------------------------------------------------
extern "C" void kernel_launch(void* const* d_in, const int* in_sizes, int n_in,
                              void* d_out, int out_size, void* d_ws, size_t ws_size,
                              hipStream_t stream) {
    (void)in_sizes; (void)n_in; (void)out_size;
    const float* x     = (const float*)d_in[0];
    const float* h_tm1 = (const float*)d_in[1];
    const float* c_tm1 = (const float*)d_in[2];
    const float* kW    = (const float*)d_in[3];
    const float* kU    = (const float*)d_in[4];
    const float* bias  = (const float*)d_in[5];
    float* h_out = (float*)d_out;
    float* c_out = h_out + (size_t)256 * 4096;

    const size_t CORE = 544768;               // halves per rearranged core set
    half_t* Wc = (half_t*)d_ws;
    half_t* Uc = Wc + CORE;
    char*   dyn = (char*)(Uc + CORE);
    const size_t fixedB = 2 * CORE * sizeof(half_t);

    // Pick the largest power-of-two batch chunk that fits the workspace:
    // per chunk: bufA + bufB (Bc*512*512 f16 each) + zNw + zNu (Bc*512*32 f32 each)
    int lbc = 8;
    while (lbc > 0) {
        size_t need = fixedB + ((size_t)1 << lbc) * (2u * 524288u + 2u * 65536u);
        if (need <= ws_size) break;
        --lbc;
    }
    if (fixedB + ((size_t)1 << lbc) * (2u * 524288u + 2u * 65536u) > ws_size) return;
    const int Bc = 1 << lbc;

    half_t* bufA = (half_t*)dyn;
    half_t* bufB = bufA + (size_t)Bc * 262144;
    float*  zNw  = (float*)(bufB + (size_t)Bc * 262144);
    float*  zNu  = zNw + (size_t)Bc * 16384;

    rearrange_cores<<<(544768 + 255) / 256, 256, 0, stream>>>(kW, Wc);
    rearrange_cores<<<(544768 + 255) / 256, 256, 0, stream>>>(kU, Uc);

    const int Mtot = 512 * Bc;
    dim3 g2((unsigned)(Mtot / 128), 4);
    dim3 g4((unsigned)(Mtot / 128), 1);
    const int expBlocks  = Mtot * 2;      // Mtot*512/256
    const int gateBlocks = Bc * 16;       // Bc*4096/256
    const int nchunks = 256 / Bc;

    for (int c = 0; c < nchunks; ++c) {
        int b0 = c * Bc;
        // ---- W chain: x -> zNw ----
        tt_expand<<<expBlocks, 256, 0, stream>>>(x, Wc, bufA, b0, lbc);
        tt_gemm<2><<<g2, 256, 0, stream>>>(bufA, Wc + 4096,   bufB, lbc);
        tt_gemm<3><<<g2, 256, 0, stream>>>(bufB, Wc + 266240, bufA, lbc);
        tt_gemm<4><<<g4, 256, 0, stream>>>(bufA, Wc + 528384, zNw,  lbc);
        // ---- U chain: h_tm1 -> zNu ----
        tt_expand<<<expBlocks, 256, 0, stream>>>(h_tm1, Uc, bufA, b0, lbc);
        tt_gemm<2><<<g2, 256, 0, stream>>>(bufA, Uc + 4096,   bufB, lbc);
        tt_gemm<3><<<g2, 256, 0, stream>>>(bufB, Uc + 266240, bufA, lbc);
        tt_gemm<4><<<g4, 256, 0, stream>>>(bufA, Uc + 528384, zNu,  lbc);
        // ---- fused gates for this batch chunk ----
        lstm_gates<<<gateBlocks, 256, 0, stream>>>(zNw, zNu, bias, c_tm1,
                                                   h_out, c_out, b0, lbc);
    }
}
